// CharBertEmbeddings_19559281066261
// MI455X (gfx1250) — matrix operand order repaired and checked by reading, not compiled, not run on hardware
//
#include <hip/hip_runtime.h>
#include <hip/hip_bf16.h>
#include <math.h>

// ---------------------------------------------------------------------------
// CharBERT char-GRU pooling for MI455X (gfx1250, wave32, WMMA).
//   k1: embedding gather + input projection (bf16 WMMA, fp32 accum)
//   k2: bidirectional GRU scan, W_hh resident in 320KB WGP LDS, 18 WMMA/step,
//       x_t streamed with GLOBAL_LOAD_ASYNC_TO_LDS_B128 (ASYNCcnt path)
//   k3: start/end gather into output
// ---------------------------------------------------------------------------

typedef __attribute__((ext_vector_type(16))) __bf16 v16bf;
typedef __attribute__((ext_vector_type(8)))  __bf16 v8bf;
typedef __attribute__((ext_vector_type(8)))  float  v8f;

#define B_   32
#define L_   2048
#define S_   512
#define E_   256
#define H_   192
#define G3H  576   // 3*H

static __device__ __forceinline__ v16bf cat8(v8bf lo, v8bf hi) {
  return __builtin_shufflevector(lo, hi, 0,1,2,3,4,5,6,7,8,9,10,11,12,13,14,15);
}
static __device__ __forceinline__ float sigmoidf_(float x) {
  return 1.0f / (1.0f + __expf(-x));
}

// Async global->LDS copy, 16 bytes per lane, tracked by ASYNCcnt.
// lds_off: LDS byte offset (generic shared-pointer low 32 bits = LDS offset).
static __device__ __forceinline__ void async_copy_b128(unsigned lds_off,
                                                       unsigned long long gaddr) {
  asm volatile("global_load_async_to_lds_b128 %0, %1, off"
               :: "v"(lds_off), "v"(gaddr) : "memory");
}
static __device__ __forceinline__ void wait_asynccnt0() {
  asm volatile("s_wait_asynccnt 0x0" ::: "memory");
}

// ---------------------------------------------------------------------------
// Kernel 1: xp[dir] = (emb[ids] @ W_ih^T + b_ih) in bf16, stored time-major
// [L][B][3H] so the GRU scan reads one contiguous slab per step.
// Block: 256 threads (8 waves). Tile: 128 rows x 64 cols, K=256.
// grid = (512 row-blocks, 2 dirs * 9 col-chunks)
// ---------------------------------------------------------------------------
__global__ __launch_bounds__(256) void embed_proj_kernel(
    const int* __restrict__ ids, const float* __restrict__ emb,
    const float* __restrict__ w_f, const float* __restrict__ b_f,
    const float* __restrict__ w_b, const float* __restrict__ b_b,
    __bf16* __restrict__ xp_f, __bf16* __restrict__ xp_b)
{
  extern __shared__ char smem1[];
  __bf16* Ash = (__bf16*)smem1;     // [128][256] bf16 activations
  __bf16* Bsh = Ash + 128 * 256;    // [64][256]  bf16 weight slice

  const int tid   = threadIdx.x;
  const int rb    = blockIdx.x;         // 0..511 (128 rows each)
  const int dir   = blockIdx.y / 9;
  const int nc    = blockIdx.y % 9;
  const int ncol0 = nc * 64;

  const float* w  = dir ? w_b  : w_f;
  const float* bi = dir ? b_b  : b_f;
  __bf16*      xp = dir ? xp_b : xp_f;

  // Stage A: gather 128 embedding rows, fp32 -> bf16. 2 threads per row.
  {
    const int row = tid >> 1, half = tid & 1;
    int id = ids[rb * 128 + row];
    if (id < 0 || id >= 1001) id = 0;
    const float4* e4 = (const float4*)(emb + (size_t)id * E_ + half * 128);
    __bf16* dst = Ash + row * 256 + half * 128;
    #pragma unroll
    for (int i = 0; i < 32; ++i) {
      float4 v = e4[i];
      dst[i*4+0] = (__bf16)v.x; dst[i*4+1] = (__bf16)v.y;
      dst[i*4+2] = (__bf16)v.z; dst[i*4+3] = (__bf16)v.w;
    }
  }
  // Stage B: 64 weight rows (output columns), coalesced.
  for (int e = tid; e < 64 * 256; e += 256)
    Bsh[e] = (__bf16)w[(size_t)ncol0 * 256 + e];
  __syncthreads();

  const int lane = tid & 31;
  const int mt   = tid >> 5;      // wave id -> M tile 0..7
  const int m    = lane & 15;
  const int kh   = lane >> 4;

  // C tiles init with bias broadcast per column (column = lane%16).
  v8f acc[4];
  #pragma unroll
  for (int nt = 0; nt < 4; ++nt) {
    const float bv = bi[ncol0 + nt * 16 + m];
    #pragma unroll
    for (int i = 0; i < 8; ++i) acc[nt][i] = bv;
  }

  #pragma unroll
  for (int kt = 0; kt < 8; ++kt) {
    // A 16x32 bf16 frag: lane holds K chunks [kh*8,+8) and [16+kh*8,+8)
    const __bf16* pa = Ash + (mt * 16 + m) * 256 + kt * 32 + kh * 8;
    const v16bf a = cat8(*(const v8bf*)pa, *(const v8bf*)(pa + 16));
    #pragma unroll
    for (int nt = 0; nt < 4; ++nt) {
      // B 32x16 bf16 frag: lane = column, K chunks [kh*8,+8) and [16+kh*8,+8)
      const __bf16* pb = Bsh + (nt * 16 + m) * 256 + kt * 32 + kh * 8;
      const v16bf b = cat8(*(const v8bf*)pb, *(const v8bf*)(pb + 16));
      acc[nt] = __builtin_amdgcn_wmma_f32_16x16x32_bf16(
          false, a, false, b, (short)0, acc[nt], false, false);
    }
  }

  // Store bf16, time-major [L][B][3H].
  #pragma unroll
  for (int nt = 0; nt < 4; ++nt) {
    const int col = ncol0 + nt * 16 + m;
    #pragma unroll
    for (int v = 0; v < 8; ++v) {
      const int r = rb * 128 + mt * 16 + kh * 8 + v;  // flat (b*L + l)
      const int b = r >> 11;
      const int l = r & 2047;
      xp[((size_t)(l * B_ + b)) * G3H + col] = (__bf16)acc[nt][v];
    }
  }
}

// ---------------------------------------------------------------------------
// Kernel 2: GRU scan. grid = 4 blocks: (dir, batch-half). 384 threads = 12
// waves; wave w owns hidden columns [w*16, w*16+16) and computes its r/z/n
// WMMA tiles (N-tiles w, w+12, w+24) so the gate math stays in registers.
// LDS: W_hh bf16 [576][192] (216KB) + double-buffered x_t (36KB) + h (12KB).
// x_t slabs stream via async global->LDS (no VGPR roundtrip); one barrier
// per step, preceded by s_wait_asynccnt 0.
// ---------------------------------------------------------------------------
__global__ __launch_bounds__(384) void gru_scan_kernel(
    const __bf16* __restrict__ xp_f, const __bf16* __restrict__ xp_b,
    const float* __restrict__ whh_f, const float* __restrict__ bhh_f,
    const float* __restrict__ whh_b, const float* __restrict__ bhh_b,
    float* __restrict__ all_h)
{
  extern __shared__ char smem2[];
  __bf16* Wsh = (__bf16*)smem2;           // [576][192]
  __bf16* XT  = Wsh + G3H * H_;           // 2 x [16][576]
  __bf16* Hb  = XT + 2 * 16 * G3H;        // 2 x [16][192]

  const int tid = threadIdx.x;
  const int dir = blockIdx.x & 1;
  const int bh  = blockIdx.x >> 1;        // batch half: rows bh*16..+16
  const __bf16* xp  = dir ? xp_b  : xp_f;
  const float*  whh = dir ? whh_b : whh_f;
  const float*  bhh = dir ? bhh_b : bhh_f;

  // LDS byte offset of XT (generic LDS pointer low 32 bits = LDS offset).
  const unsigned xtBase = (unsigned)(unsigned long long)(uintptr_t)(void*)XT;
  const unsigned slabBytes = 16 * G3H * 2;          // 18432 B per buffer

  for (int i = tid; i < G3H * H_; i += 384) Wsh[i] = (__bf16)whh[i];
  for (int i = tid; i < 16 * H_;  i += 384) Hb[i]  = (__bf16)0.0f;
  // Async prefetch of step 0's x_t slab into buffer 0.
  {
    const int t0 = dir ? (L_ - 1) : 0;
    const unsigned long long src =
        (unsigned long long)(uintptr_t)(xp + ((size_t)(t0 * B_ + bh * 16)) * G3H);
    #pragma unroll
    for (int k = 0; k < 3; ++k) {
      const int i = tid + k * 384;                  // 1152 x 16B
      async_copy_b128(xtBase + (unsigned)i * 16u, src + (unsigned long long)i * 16u);
    }
  }
  wait_asynccnt0();
  __syncthreads();

  const int lane = tid & 31;
  const int wv   = tid >> 5;          // 0..11 -> column group
  const int nl   = lane & 15;
  const int kh   = lane >> 4;
  const int hc   = wv * 16 + nl;      // this lane's hidden column

  const float br = bhh[hc];
  const float bz = bhh[H_ + hc];
  const float bn = bhh[2 * H_ + hc];

  for (int s = 0; s < L_; ++s) {
    const int cur = s & 1, nxt = cur ^ 1;
    const int t = dir ? (L_ - 1 - s) : s;

    // Issue async prefetch of next step's x_t slab into the other buffer;
    // it drains (ASYNCcnt) while the WMMA chain and gate math execute.
    {
      const int sn = (s + 1 < L_) ? (s + 1) : s;
      const int tn = dir ? (L_ - 1 - sn) : sn;
      const unsigned long long src =
          (unsigned long long)(uintptr_t)(xp + ((size_t)(tn * B_ + bh * 16)) * G3H);
      const unsigned dstBase = xtBase + (unsigned)nxt * slabBytes;
      #pragma unroll
      for (int k = 0; k < 3; ++k) {
        const int i = tid + k * 384;
        async_copy_b128(dstBase + (unsigned)i * 16u, src + (unsigned long long)i * 16u);
      }
    }

    const __bf16* hb = Hb + cur * 16 * H_;
    // A frags for h[16 x 192]: one per K tile, shared by r/z/n tiles.
    v16bf A[6];
    #pragma unroll
    for (int kt = 0; kt < 6; ++kt) {
      const __bf16* p = hb + nl * H_ + kt * 32 + kh * 8;
      A[kt] = cat8(*(const v8bf*)p, *(const v8bf*)(p + 16));
    }

    v8f cr, cz, cn;
    #pragma unroll
    for (int i = 0; i < 8; ++i) { cr[i] = br; cz[i] = bz; cn[i] = bn; }

    #pragma unroll
    for (int kt = 0; kt < 6; ++kt) {
      const __bf16* prow = Wsh + (size_t)hc * H_ + kt * 32 + kh * 8;
      const v16bf bfr = cat8(*(const v8bf*)prow, *(const v8bf*)(prow + 16)); // r rows
      cr = __builtin_amdgcn_wmma_f32_16x16x32_bf16(
          false, A[kt], false, bfr, (short)0, cr, false, false);
      const v16bf bfz = cat8(*(const v8bf*)(prow + (size_t)H_ * H_), *(const v8bf*)(prow + (size_t)H_ * H_ + 16));   // z rows
      cz = __builtin_amdgcn_wmma_f32_16x16x32_bf16(
          false, A[kt], false, bfz, (short)0, cz, false, false);
      const v16bf bfn = cat8(*(const v8bf*)(prow + (size_t)2 * H_ * H_), *(const v8bf*)(prow + (size_t)2 * H_ * H_ + 16)); // n rows
      cn = __builtin_amdgcn_wmma_f32_16x16x32_bf16(
          false, A[kt], false, bfn, (short)0, cn, false, false);
    }

    // Gate math on accumulator elements: row b = kh*8+v, col = hc.
    const __bf16* xt = XT + cur * 16 * G3H;
    __bf16* hn = Hb + nxt * 16 * H_;
    #pragma unroll
    for (int v = 0; v < 8; ++v) {
      const int bl = kh * 8 + v;
      const float xr   = (float)xt[bl * G3H + hc];
      const float xz   = (float)xt[bl * G3H + H_ + hc];
      const float xn   = (float)xt[bl * G3H + 2 * H_ + hc];
      const float hold = (float)hb[bl * H_ + hc];
      const float r  = sigmoidf_(xr + cr[v]);
      const float z  = sigmoidf_(xz + cz[v]);
      const float g  = tanhf(xn + r * cn[v]);
      const float hv = (1.0f - z) * g + z * hold;
      hn[bl * H_ + hc] = (__bf16)hv;
      const int bg = bh * 16 + bl;
      all_h[((size_t)bg * L_ + t) * (2 * H_) + dir * H_ + hc] = hv;
    }
    wait_asynccnt0();       // this wave's prefetch landed in LDS
    __syncthreads();        // everyone's landed -> safe to consume next step
  }
}

// ---------------------------------------------------------------------------
// Kernel 3: out[b,s] = concat(all_h[b, start[b,s]], all_h[b, end[b,s]])
// ---------------------------------------------------------------------------
__global__ __launch_bounds__(128) void gather_kernel(
    const int* __restrict__ start_ids, const int* __restrict__ end_ids,
    const float* __restrict__ all_h, float* __restrict__ out)
{
  const int bs = blockIdx.x;            // b*512 + s
  const int b  = bs >> 9;
  int t0 = start_ids[bs], t1 = end_ids[bs];
  if (t0 < 0) t0 = 0; if (t0 >= L_) t0 = L_ - 1;
  if (t1 < 0) t1 = 0; if (t1 >= L_) t1 = L_ - 1;
  const float* h0 = all_h + ((size_t)b * L_ + t0) * (2 * H_);
  const float* h1 = all_h + ((size_t)b * L_ + t1) * (2 * H_);
  float* o = out + (size_t)bs * 768;
  for (int i = threadIdx.x; i < 2 * H_; i += 128) {
    o[i]          = h0[i];
    o[2 * H_ + i] = h1[i];
  }
}

// ---------------------------------------------------------------------------
extern "C" void kernel_launch(void* const* d_in, const int* in_sizes, int n_in,
                              void* d_out, int out_size, void* d_ws, size_t ws_size,
                              hipStream_t stream) {
  const int*   ids   = (const int*)  d_in[0];
  const int*   sid   = (const int*)  d_in[1];
  const int*   eid   = (const int*)  d_in[2];
  const float* emb   = (const float*)d_in[3];
  const float* wih_f = (const float*)d_in[4];
  const float* whh_f = (const float*)d_in[5];
  const float* bih_f = (const float*)d_in[6];
  const float* bhh_f = (const float*)d_in[7];
  const float* wih_b = (const float*)d_in[8];
  const float* whh_b = (const float*)d_in[9];
  const float* bih_b = (const float*)d_in[10];
  const float* bhh_b = (const float*)d_in[11];

  const size_t xpElems = (size_t)L_ * B_ * G3H;   // per direction
  __bf16* xp_f  = (__bf16*)d_ws;
  __bf16* xp_b  = xp_f + xpElems;
  float*  all_h = (float*)(xp_b + xpElems);       // [B][L][2H] fp32

  const size_t lds1 = (size_t)(128 * 256 + 64 * 256) * sizeof(__bf16); // 96 KB
  embed_proj_kernel<<<dim3(512, 18), 256, lds1, stream>>>(
      ids, emb, wih_f, bih_f, wih_b, bih_b, xp_f, xp_b);

  const size_t lds2 =
      (size_t)(G3H * H_ + 2 * 16 * G3H + 2 * 16 * H_) * sizeof(__bf16); // 264 KB
  gru_scan_kernel<<<4, 384, lds2, stream>>>(
      xp_f, xp_b, whh_f, bhh_f, whh_b, bhh_b, all_h);

  gather_kernel<<<B_ * S_, 128, 0, stream>>>(sid, eid, all_h, (float*)d_out);
}
